// AttentionLayer_46583215292607
// MI455X (gfx1250) — matrix operand order repaired and checked by reading, not compiled, not run on hardware
//
#include <hip/hip_runtime.h>
#include <math.h>

// CDNA5 / gfx1250. wave32. Bahdanau additive attention, bf16 WMMA core.
// B=32, T=4096, Dh=Ds=1024, Da=512.

typedef __attribute__((ext_vector_type(16))) __bf16 bf16x16;
typedef __attribute__((ext_vector_type(8)))  __bf16 bf16x8;
typedef __attribute__((ext_vector_type(8)))  float  f32x8;

#define B_  32
#define T_  4096
#define DH  1024
#define DS_ 1024
#define DA  512

// Branch-free tanh: prefer gfx1250 hardware v_tanh_f32 if the builtin exists,
// else exp-based approximation (v_exp_f32 + v_rcp_f32, no EXEC divergence).
__device__ __forceinline__ float fast_tanh(float x) {
#if __has_builtin(__builtin_amdgcn_tanhf)
  return __builtin_amdgcn_tanhf(x);
#else
  float ax = __builtin_fabsf(x);
  float ex = __expf(2.0f * ax);
  float t  = 1.0f - 2.0f * __builtin_amdgcn_rcpf(ex + 1.0f);
  return __builtin_copysignf(t, x);
#endif
}

// ---------------------------------------------------------------------------
// Kernel 1: U_t[n][k] = bf16(U_a[k][n])  (transpose+cast; 1 MiB, L2-resident)
// ---------------------------------------------------------------------------
__global__ __launch_bounds__(256) void k_transpose_cast(
    const float* __restrict__ U, __bf16* __restrict__ Ut) {
  int idx = blockIdx.x * 256 + threadIdx.x;   // 0 .. DA*DH-1, output-linear
  int n = idx >> 10;                          // / DH
  int k = idx & (DH - 1);
  Ut[idx] = (__bf16)U[(size_t)k * DA + n];
}

// ---------------------------------------------------------------------------
// Kernel 2: Ws[b][n] = sum_k s[b][k] * W_a[k][n]
// ---------------------------------------------------------------------------
__global__ __launch_bounds__(256) void k_ws(
    const float* __restrict__ s, const float* __restrict__ W,
    float* __restrict__ Ws) {
  int b = blockIdx.y;
  int n = blockIdx.x * 256 + threadIdx.x;
  const float* sb = s + (size_t)b * DS_;
  float acc = 0.f;
  for (int k = 0; k < DS_; ++k)
    acc = fmaf(sb[k], W[(size_t)k * DA + n], acc);
  Ws[b * DA + n] = acc;
}

// ---------------------------------------------------------------------------
// Kernel 3: scores e[b][t] = v_a . tanh(Ws[b] + h[b,t,:] @ U_a)
//   One block = one (b, 16-row t-tile). 128 threads = 4 waves.
//   h tile staged to LDS as bf16 once; 4 waves split the 32 N-chunks of Da.
//   Dual accumulators (even/odd K-chunks) for two independent WMMA chains;
//   both seeded with Ws/2 so their sum carries the bias exactly once... no:
//   acc0 seeded with Ws, acc1 with 0; summed before tanh.
// ---------------------------------------------------------------------------
#define TILE_T 16
#define PAD    8
#define LDSTR  (DH + PAD)   // bf16 elements per LDS row (bank-spread)

__global__ __launch_bounds__(128) void k_scores(
    const float* __restrict__ h, const __bf16* __restrict__ Ut,
    const float* __restrict__ Ws, const float* __restrict__ va,
    float* __restrict__ e) {
  __shared__ __bf16 sh[TILE_T * LDSTR];
  __shared__ float  se[TILE_T];

  const int tid  = threadIdx.x;
  const int lane = tid & 31;
  const int wave = tid >> 5;                       // 0..3
  const int tilesPerB = T_ / TILE_T;               // 256
  const int b  = blockIdx.x / tilesPerB;
  const int t0 = (blockIdx.x % tilesPerB) * TILE_T;

  // ---- stage h[b, t0:t0+16, 0:1024] -> LDS bf16 (64 KiB f32 -> 32 KiB) ----
  const float* hb = h + (size_t)(b * T_ + t0) * DH;
  #pragma unroll 4
  for (int it = 0; it < (TILE_T * DH / 4) / 128; ++it) {
    int idx = it * 128 + tid;                      // float4 index
    int row = idx >> 8;                            // 256 float4 per row
    int c4  = (idx & 255) << 2;
    float4 v = *(const float4*)(hb + (size_t)row * DH + c4);
    __bf16* dst = sh + row * LDSTR + c4;
    dst[0] = (__bf16)v.x; dst[1] = (__bf16)v.y;
    dst[2] = (__bf16)v.z; dst[3] = (__bf16)v.w;
  }
  if (tid < TILE_T) se[tid] = 0.f;
  __syncthreads();

  // ---- per-lane fragment geometry (16-bit WMMA layouts, wave32) ----
  const int lrow   = lane & 15;                    // A row / B col / C col
  const int khalfA = (lane >> 4) << 3;             // A: K base 0 or 8
  const int khalfB = (lane >> 4) << 3;             // B: K base 0 or 8
  const __bf16* aptr = sh + lrow * LDSTR + khalfA;

  float r[8];
  #pragma unroll
  for (int i = 0; i < 8; ++i) r[i] = 0.f;

  for (int nc = wave; nc < DA / 16; nc += 4) {     // 8 N-chunks per wave
    const int n  = nc * 16 + lrow;
    const float wsv = Ws[b * DA + n];
    const float vav = va[n];
    f32x8 acc0, acc1;
    #pragma unroll
    for (int i = 0; i < 8; ++i) { acc0[i] = wsv; acc1[i] = 0.f; }

    const __bf16* bptr = Ut + (size_t)n * DH + khalfB;

    #pragma unroll 2
    for (int kc2 = 0; kc2 < DH / 64; ++kc2) {      // 2 WMMAs / iter, 2 chains
      #pragma unroll
      for (int half = 0; half < 2; ++half) {
        const int kc = kc2 * 2 + half;
        bf16x8 a0 = *(const bf16x8*)(aptr + kc * 32);        // K +0..7
        bf16x8 a1 = *(const bf16x8*)(aptr + kc * 32 + 16);   // K +16..23
        bf16x8 b0 = *(const bf16x8*)(bptr + kc * 32);        // K +0..7
        bf16x8 b1 = *(const bf16x8*)(bptr + kc * 32 + 16);   // K +16..23
        bf16x16 afrag, bfrag;
        #pragma unroll
        for (int i = 0; i < 8; ++i) {
          afrag[i] = a0[i]; afrag[i + 8] = a1[i];
          bfrag[i] = b0[i]; bfrag[i + 8] = b1[i];
        }
        if (half == 0)
          acc0 = __builtin_amdgcn_wmma_f32_16x16x32_bf16(
              false, afrag, false, bfrag, (short)0, acc0, false, false);
        else
          acc1 = __builtin_amdgcn_wmma_f32_16x16x32_bf16(
              false, afrag, false, bfrag, (short)0, acc1, false, false);
      }
    }
    // acc = Ws + Uh at (row m = v + 8*(lane/16), col n); fold tanh * v_a
    #pragma unroll
    for (int i = 0; i < 8; ++i)
      r[i] += fast_tanh(acc0[i] + acc1[i]) * vav;
  }

  // reduce over the 16 columns held across each 16-lane half
  #pragma unroll
  for (int off = 1; off < 16; off <<= 1) {
    #pragma unroll
    for (int i = 0; i < 8; ++i) r[i] += __shfl_xor(r[i], off, 32);
  }
  if ((lane & 15) == 0) {
    const int rbase = (lane >> 4) * 8;             // lane0 -> rows 0..7, lane16 -> 8..15
    #pragma unroll
    for (int i = 0; i < 8; ++i) atomicAdd(&se[rbase + i], r[i]);
  }
  __syncthreads();
  if (tid < TILE_T) e[b * T_ + t0 + tid] = se[tid];
}

// ---------------------------------------------------------------------------
// Kernel 4: in-place stable softmax over T per batch row
// ---------------------------------------------------------------------------
__global__ __launch_bounds__(256) void k_softmax(float* __restrict__ e) {
  __shared__ float red[256];
  const int b = blockIdx.x, tid = threadIdx.x;
  float* eb = e + b * T_;

  float m = -3.402823466e38f;
  for (int t = tid; t < T_; t += 256) m = fmaxf(m, eb[t]);
  red[tid] = m; __syncthreads();
  for (int s2 = 128; s2 > 0; s2 >>= 1) {
    if (tid < s2) red[tid] = fmaxf(red[tid], red[tid + s2]);
    __syncthreads();
  }
  m = red[0]; __syncthreads();

  float sum = 0.f;
  for (int t = tid; t < T_; t += 256) {
    float x = __expf(eb[t] - m);
    eb[t] = x;
    sum += x;
  }
  red[tid] = sum; __syncthreads();
  for (int s2 = 128; s2 > 0; s2 >>= 1) {
    if (tid < s2) red[tid] += red[tid + s2];
    __syncthreads();
  }
  const float inv = 1.0f / red[0];
  for (int t = tid; t < T_; t += 256) eb[t] *= inv;
}

// ---------------------------------------------------------------------------
// Kernel 5: context c[b][dh] = sum_t a[b][t] * h[b][t][dh]  (HBM-bound)
// ---------------------------------------------------------------------------
__global__ __launch_bounds__(256) void k_context(
    const float* __restrict__ a, const float* __restrict__ h,
    float* __restrict__ c) {
  const int b  = blockIdx.y;
  const int dh = blockIdx.x * 256 + threadIdx.x;
  const float* ab = a + b * T_;
  const float* hb = h + (size_t)b * T_ * DH + dh;
  float acc = 0.f;
  #pragma unroll 4
  for (int t = 0; t < T_; ++t)
    acc = fmaf(ab[t], hb[(size_t)t * DH], acc);   // ab[t] uniform -> scalarized
  c[b * DH + dh] = acc;
}

// ---------------------------------------------------------------------------
extern "C" void kernel_launch(void* const* d_in, const int* in_sizes, int n_in,
                              void* d_out, int out_size, void* d_ws, size_t ws_size,
                              hipStream_t stream) {
  const float* s  = (const float*)d_in[0];   // [32,1024]
  const float* h  = (const float*)d_in[1];   // [32,4096,1024]
  const float* Wa = (const float*)d_in[2];   // [1024,512]
  const float* Ua = (const float*)d_in[3];   // [1024,512]
  const float* va = (const float*)d_in[4];   // [512]
  float* c = (float*)d_out;                  // [32,1024]

  // workspace layout (~1.6 MiB): Ut bf16 | Ws f32 | e f32
  char* ws = (char*)d_ws;
  __bf16* Ut  = (__bf16*)ws;                             // 512*1024*2 = 1 MiB
  float*  Wsb = (float*)(ws + (1u << 20));               // 32*512*4  = 64 KiB
  float*  e   = (float*)(ws + (1u << 20) + (1u << 16));  // 32*4096*4 = 512 KiB

  k_transpose_cast<<<(DA * DH) / 256, 256, 0, stream>>>(Ua, Ut);
  k_ws<<<dim3(DA / 256, B_), 256, 0, stream>>>(s, Wa, Wsb);
  k_scores<<<B_ * (T_ / TILE_T), 128, 0, stream>>>(h, Ut, Wsb, va, e);
  k_softmax<<<B_, 256, 0, stream>>>(e);
  k_context<<<dim3(DH / 256, B_), 256, 0, stream>>>(e, h, c);
}
